// LTAttention_6880537608676
// MI455X (gfx1250) — matrix operand order repaired and checked by reading, not compiled, not run on hardware
//
#include <hip/hip_runtime.h>
#include <math.h>

// ---------------------------------------------------------------------------
// Types / helpers
// ---------------------------------------------------------------------------
typedef unsigned short u16;
typedef __attribute__((ext_vector_type(16))) __bf16 v16bf;
typedef __attribute__((ext_vector_type(8)))  float  v8f;
typedef __attribute__((ext_vector_type(4)))  unsigned int u32x4;
typedef __attribute__((ext_vector_type(8)))  int i32x8;
typedef __attribute__((ext_vector_type(4)))  int i32x4;

union FragU { v16bf v; uint4 q[2]; };

__device__ __forceinline__ u16 f2bf(float x) {
  union { float f; unsigned u; } c; c.f = x;
  unsigned r = c.u + 0x7FFFu + ((c.u >> 16) & 1u);   // round-to-nearest-even
  return (u16)(r >> 16);
}
__device__ __forceinline__ float bf2f(u16 x) {
  union { unsigned u; float f; } c; c.u = ((unsigned)x) << 16;
  return c.f;
}

// A-matrix fragment (16xK tile slice, K-step 32) from a row-major [rows, ld]
// bf16 buffer. CDNA5 layout: lanes 0-15 hold row M=lane, K={0..7,16..23};
// lanes 16-31 hold row M=lane-16, K={8..15,24..31}.
__device__ __forceinline__ v16bf frag_a(const u16* p0, int ld, int lane) {
  const int r = lane & 15, h = lane >> 4;
  const u16* p = p0 + r * ld + h * 8;
  FragU f;
  f.q[0] = *(const uint4*)(p);        // K = h*8 + 0..7
  f.q[1] = *(const uint4*)(p + 16);   // K = h*8 + 16..23
  return f.v;
}
// B-matrix fragment (Kx16) from a row-major [N, ld] buffer whose rows are the
// 16 output columns and whose last dim is the contraction dim.
// Layout: lanes 0-15 = col N=lane with K={0..7,16..23}; lanes 16-31 = K={8..15,24..31}.
__device__ __forceinline__ v16bf frag_b(const u16* p0, int ld, int lane) {
  const int nn = lane & 15, h = lane >> 4;
  const u16* p = p0 + nn * ld + h * 8;
  FragU f;
  f.q[0] = *(const uint4*)(p);
  f.q[1] = *(const uint4*)(p + 16);
  return f.v;
}

#define WMMA_BF16(a, b, c) \
  __builtin_amdgcn_wmma_f32_16x16x32_bf16(false, (a), false, (b), (short)0, (c), false, false)

// ---------------------------------------------------------------------------
// TDM: async 2D bf16 tile copy (global, row-major, strideElems between rows)
// -> packed LDS at byte offset lds_off. Descriptor per CDNA5 ISA ch.8:
//   group0: [1:0]=count=1, [63:32]=lds_addr, [120:64]=global_addr, [127:126]=2
//   group1: [17:16]=data_size(1=2B), [79:48]=tensor_dim0, [111:80]=tensor_dim1,
//           [127:112]=tile_dim0, [143:128]=tile_dim1, [207:160]=dim0_stride
// Tensor dims set equal to tile dims (tile origin = global_addr, no OOB).
// Groups 2/3 (and the trailing descriptor group) zero for <=2D tensors.
// Issue from one wave; wait TENSORcnt.
// ---------------------------------------------------------------------------
__device__ __forceinline__ void tdm_load_2d(unsigned lds_off, const void* gptr,
                                            unsigned tileC, unsigned tileR,
                                            unsigned strideElems) {
  const unsigned long long ga = (unsigned long long)gptr;
  u32x4 g0;
  g0[0] = 1u;                                            // count=1, user mode
  g0[1] = lds_off;                                       // LDS byte address
  g0[2] = (unsigned)(ga & 0xFFFFFFFFu);                  // global_addr[31:0]
  g0[3] = (unsigned)((ga >> 32) & 0x01FFFFFFu) | (2u << 30);  // addr[56:32], type=2
  i32x8 g1;
  g1[0] = (int)(1u << 16);                               // data_size=2B, wg_mask=0
  g1[1] = (int)((tileC & 0xFFFFu) << 16);                // tensor_dim0[15:0]
  g1[2] = (int)(((tileC >> 16) & 0xFFFFu) |              // tensor_dim0[31:16]
                ((tileR & 0xFFFFu) << 16));              // tensor_dim1[15:0]
  g1[3] = (int)(((tileR >> 16) & 0xFFFFu) |              // tensor_dim1[31:16]
                ((tileC & 0xFFFFu) << 16));              // tile_dim0
  g1[4] = (int)(tileR & 0xFFFFu);                        // tile_dim1 (tile_dim2=0)
  g1[5] = (int)strideElems;                              // tensor_dim0_stride[31:0]
  g1[6] = 0;                                             // stride hi bits
  g1[7] = 0;
  const i32x4 z4 = {0, 0, 0, 0};
  const i32x8 z8 = {0, 0, 0, 0, 0, 0, 0, 0};
  __builtin_amdgcn_tensor_load_to_lds(g0, g1, z4, z4, z8, 0);
}

// ---------------------------------------------------------------------------
// Kernel: f32 -> bf16 convert (grid stride)
// ---------------------------------------------------------------------------
__global__ __launch_bounds__(256) void f32_to_bf16_k(const float* __restrict__ in,
                                                     u16* __restrict__ out, int n) {
  int i = blockIdx.x * 256 + threadIdx.x;
  const int stride = gridDim.x * 256;
  for (; i < n; i += stride) out[i] = f2bf(in[i]);
}

// ---------------------------------------------------------------------------
// Kernel: C[M,N] = A[M,K] @ B[N,K]^T + bias[N]   (bf16 in, f32 accum)
// 128x128 block tile, 8 waves each computing 64x32, K-step 32.
// ---------------------------------------------------------------------------
template <bool BF_OUT>
__global__ __launch_bounds__(256) void gemm_bf16_k(
    const u16* __restrict__ A, const u16* __restrict__ Bw,
    const float* __restrict__ bias, void* __restrict__ Cout,
    int M, int N, int K) {
  __shared__ u16 As[128 * 32];
  __shared__ u16 Bs[128 * 32];

  const int t = threadIdx.x, lane = t & 31, w = t >> 5;
  const int m0 = blockIdx.y * 128, n0 = blockIdx.x * 128;
  const int wm = (w & 1) * 64, wn = (w >> 1) * 32;

  const v8f zero8 = {0.f, 0.f, 0.f, 0.f, 0.f, 0.f, 0.f, 0.f};
  v8f acc[4][2];
#pragma unroll
  for (int i = 0; i < 4; ++i)
#pragma unroll
    for (int j = 0; j < 2; ++j) acc[i][j] = zero8;

  const int srow = t >> 1, scol = (t & 1) * 16;
  const u16* ga = A + (size_t)(m0 + srow) * K + scol;
  const u16* gb = Bw + (size_t)(n0 + srow) * K + scol;

  for (int k0 = 0; k0 < K; k0 += 32) {
    const uint4 a0 = *(const uint4*)(ga + k0);
    const uint4 a1 = *(const uint4*)(ga + k0 + 8);
    const uint4 b0 = *(const uint4*)(gb + k0);
    const uint4 b1 = *(const uint4*)(gb + k0 + 8);
    __builtin_prefetch(ga + k0 + 32, 0, 1);   // global_prefetch_b8 (next K tile)
    __builtin_prefetch(gb + k0 + 32, 0, 1);

    __syncthreads();   // previous iteration done reading LDS
    *(uint4*)(As + srow * 32 + scol)     = a0;
    *(uint4*)(As + srow * 32 + scol + 8) = a1;
    *(uint4*)(Bs + srow * 32 + scol)     = b0;
    *(uint4*)(Bs + srow * 32 + scol + 8) = b1;
    __syncthreads();

    v16bf af[4];
#pragma unroll
    for (int i = 0; i < 4; ++i) af[i] = frag_a(As + (wm + 16 * i) * 32, 32, lane);
#pragma unroll
    for (int j = 0; j < 2; ++j) {
      const v16bf bf = frag_b(Bs + (wn + 16 * j) * 32, 32, lane);
#pragma unroll
      for (int i = 0; i < 4; ++i) acc[i][j] = WMMA_BF16(af[i], bf, acc[i][j]);
    }
  }

  // Epilogue. C/D layout: lanes 0-15 cols, VGPR e = row mb+e.
  const int nn = lane & 15, mb = (lane >> 4) * 8;
#pragma unroll
  for (int i = 0; i < 4; ++i)
#pragma unroll
    for (int j = 0; j < 2; ++j) {
      const int gr = m0 + wm + 16 * i + mb;
      const int gc = n0 + wn + 16 * j + nn;
      const float bv = bias[gc];
#pragma unroll
      for (int e = 0; e < 8; ++e) {
        const float vv = acc[i][j][e] + bv;
        if (BF_OUT) ((u16*)Cout)[(size_t)(gr + e) * N + gc] = f2bf(vv);
        else        ((float*)Cout)[(size_t)(gr + e) * N + gc] = vv;
      }
    }
}

// ---------------------------------------------------------------------------
// Kernel: RMSNorm(q), RMSNorm(k), split qkv -> Q/K/V in [B,H,S,d] bf16 layout.
// One block per token (b,s). D = 2048, H = 16, d = 128.
// ---------------------------------------------------------------------------
__global__ __launch_bounds__(256) void rmsnorm_split_k(
    const u16* __restrict__ qkv, const float* __restrict__ qw,
    const float* __restrict__ kw, u16* __restrict__ Qb, u16* __restrict__ Kb,
    u16* __restrict__ Vb, int S, int D) {
  __shared__ float rq_s[8], rk_s[8], tot[2];
  const int bs = blockIdx.x, b = bs / S, s = bs % S;
  const u16* rowp = qkv + (size_t)bs * 3 * D;
  const int t = threadIdx.x, lane = t & 31, w = t >> 5;

  float qv[8], kv[8];
  float sq = 0.f, sk = 0.f;
#pragma unroll
  for (int i = 0; i < 8; ++i) {
    const int d = t + 256 * i;
    qv[i] = bf2f(rowp[d]);
    kv[i] = bf2f(rowp[D + d]);
    sq += qv[i] * qv[i];
    sk += kv[i] * kv[i];
  }
#pragma unroll
  for (int msk = 16; msk >= 1; msk >>= 1) {
    sq += __shfl_xor(sq, msk, 32);
    sk += __shfl_xor(sk, msk, 32);
  }
  if (lane == 0) { rq_s[w] = sq; rk_s[w] = sk; }
  __syncthreads();
  if (t == 0) {
    float a = 0.f, c = 0.f;
#pragma unroll
    for (int i = 0; i < 8; ++i) { a += rq_s[i]; c += rk_s[i]; }
    tot[0] = a; tot[1] = c;
  }
  __syncthreads();
  const float rq = rsqrtf(tot[0] / (float)D + 1e-6f);
  const float rk = rsqrtf(tot[1] / (float)D + 1e-6f);

#pragma unroll
  for (int i = 0; i < 8; ++i) {
    const int d = t + 256 * i;
    const int h = d >> 7, dd = d & 127;
    const size_t o = ((size_t)(b * 16 + h) * S + s) * 128 + dd;
    Qb[o] = f2bf(qv[i] * rq * qw[d]);
    Kb[o] = f2bf(kv[i] * rk * kw[d]);
    Vb[o] = rowp[2 * D + d];   // v: passthrough bf16
  }
}

// ---------------------------------------------------------------------------
// Kernel: flash attention. One block = 128 query rows of one (b,h).
// 8 waves; wave w owns query rows 16w..16w+15 (softmax is wave-local).
// Key blocks of 64; online softmax; output written to ctx [B*S, D] bf16.
// Q and K tiles staged via the Tensor Data Mover (wave 0 issues the DMA,
// waits TENSORcnt, workgroup barrier publishes); V is transposed on the
// vector path concurrently with the K-tile DMA.
// LDS byte map (dynamic): Qs @ 0 (32KB), Ks @ 32768 (16KB),
//                         VT @ 49152 (16KB), Ps @ 65536 (16KB).
// ---------------------------------------------------------------------------
__global__ __launch_bounds__(256) void attn_k(
    const u16* __restrict__ Qb, const u16* __restrict__ Kb,
    const u16* __restrict__ Vb, const float* __restrict__ bias,
    u16* __restrict__ ctx, int S, float scale) {
  extern __shared__ u16 sm[];
  u16* Qs = sm;                 // 128 x 128 (queries x d)
  u16* Ks = Qs + 128 * 128;     //  64 x 128 (keys x d)
  u16* VT = Ks + 64 * 128;      // 128 x 64  (d x keys, transposed)
  u16* Ps = VT + 128 * 64;      // 128 x 64  (probabilities, bf16)

  const int t = threadIdx.x, lane = t & 31, w = t >> 5;
  const int nn = lane & 15, mb = (lane >> 4) * 8;
  const int bh = blockIdx.y, b = bh >> 4, h = bh & 15;
  const int q0 = blockIdx.x * 128;

  // Stage Q tile once via TDM (wave 0 only).
  const u16* Qg = Qb + ((size_t)bh * S + q0) * 128;
  if (w == 0) {
    tdm_load_2d(/*lds_off=*/0u, Qg, /*tileC=*/128u, /*tileR=*/128u, /*stride=*/128u);
  }

  const v8f zero8 = {0.f, 0.f, 0.f, 0.f, 0.f, 0.f, 0.f, 0.f};
  float m_i[8], l_i[8];
  v8f O[8];
#pragma unroll
  for (int e = 0; e < 8; ++e) { m_i[e] = -3.0e38f; l_i[e] = 0.f; }
#pragma unroll
  for (int d = 0; d < 8; ++d) O[d] = zero8;

  for (int j0 = 0; j0 < q0 + 128; j0 += 64) {
    const u16* Kg = Kb + ((size_t)bh * S + j0) * 128;
    const u16* Vg = Vb + ((size_t)bh * S + j0) * 128;
    __syncthreads();   // previous iteration finished with Ks/VT

    // K tile: async TDM copy (64 rows x 128 cols, packed into Ks).
    if (w == 0) {
      tdm_load_2d(/*lds_off=*/128u * 128u * 2u, Kg, 128u, 64u, 128u);
    }
    // V tile: manual transposed staging (overlaps the K DMA).
    {
      const int r = t >> 2, c = (t & 3) * 32;
#pragma unroll
      for (int i = 0; i < 4; ++i) {        // V staged transposed: VT[d][key]
        uint4 vv = *(const uint4*)(Vg + r * 128 + c + i * 8);
        const u16* e8 = (const u16*)&vv;
#pragma unroll
        for (int j = 0; j < 8; ++j) VT[(c + i * 8 + j) * 64 + r] = e8[j];
      }
    }
    if (w == 0) __builtin_amdgcn_s_wait_tensorcnt(0);   // Q (first iter) + K done
    __syncthreads();

    // --- scores: S = Q @ K^T (wave rows 16w.., 64 keys) ---
    v8f sc[4];
#pragma unroll
    for (int tt = 0; tt < 4; ++tt) sc[tt] = zero8;
    v16bf aq[4];
#pragma unroll
    for (int kk = 0; kk < 4; ++kk) aq[kk] = frag_a(Qs + (16 * w) * 128 + kk * 32, 128, lane);
#pragma unroll
    for (int tt = 0; tt < 4; ++tt)
#pragma unroll
      for (int kk = 0; kk < 4; ++kk) {
        const v16bf bk = frag_b(Ks + (16 * tt) * 128 + kk * 32, 128, lane);
        sc[tt] = WMMA_BF16(aq[kk], bk, sc[tt]);
      }

    // --- scale + bias + causal mask, online softmax ---
    float mnew[8];
#pragma unroll
    for (int e = 0; e < 8; ++e) mnew[e] = m_i[e];
    const size_t brow = (size_t)h * S;
#pragma unroll
    for (int tt = 0; tt < 4; ++tt)
#pragma unroll
      for (int e = 0; e < 8; ++e) {
        const int row = q0 + 16 * w + mb + e;
        const int col = j0 + 16 * tt + nn;
        float sv = sc[tt][e] * scale + bias[(brow + row) * S + col];
        if (col > row) sv = -3.0e38f;
        sc[tt][e] = sv;
        mnew[e] = fmaxf(mnew[e], sv);
      }
#pragma unroll
    for (int msk = 8; msk >= 1; msk >>= 1)
#pragma unroll
      for (int e = 0; e < 8; ++e)
        mnew[e] = fmaxf(mnew[e], __shfl_xor(mnew[e], msk, 32));

    float alpha[8], ls[8];
#pragma unroll
    for (int e = 0; e < 8; ++e) {
      alpha[e] = __expf(m_i[e] - mnew[e]);
      m_i[e] = mnew[e];
      ls[e] = 0.f;
    }
#pragma unroll
    for (int tt = 0; tt < 4; ++tt)
#pragma unroll
      for (int e = 0; e < 8; ++e) {
        const float p = __expf(sc[tt][e] - mnew[e]);
        ls[e] += p;
        Ps[(16 * w + mb + e) * 64 + 16 * tt + nn] = f2bf(p);  // wave-local strip
      }
#pragma unroll
    for (int msk = 8; msk >= 1; msk >>= 1)
#pragma unroll
      for (int e = 0; e < 8; ++e) ls[e] += __shfl_xor(ls[e], msk, 32);
#pragma unroll
    for (int e = 0; e < 8; ++e) l_i[e] = l_i[e] * alpha[e] + ls[e];
#pragma unroll
    for (int d = 0; d < 8; ++d)
#pragma unroll
      for (int e = 0; e < 8; ++e) O[d][e] *= alpha[e];

    // --- O += P @ V ---
    v16bf ap[2];
#pragma unroll
    for (int kk = 0; kk < 2; ++kk) ap[kk] = frag_a(Ps + (16 * w) * 64 + kk * 32, 64, lane);
#pragma unroll
    for (int d = 0; d < 8; ++d)
#pragma unroll
      for (int kk = 0; kk < 2; ++kk) {
        const v16bf bv = frag_b(VT + (16 * d) * 64 + kk * 32, 64, lane);
        O[d] = WMMA_BF16(ap[kk], bv, O[d]);
      }
  }

  // Normalize and emit ctx[b*S + s, h*128 + d] (bf16).
#pragma unroll
  for (int e = 0; e < 8; ++e) l_i[e] = 1.0f / l_i[e];
#pragma unroll
  for (int d = 0; d < 8; ++d)
#pragma unroll
    for (int e = 0; e < 8; ++e) {
      const int row = q0 + 16 * w + mb + e;
      ctx[((size_t)b * S + row) * 2048 + h * 128 + 16 * d + nn] = f2bf(O[d][e] * l_i[e]);
    }
}

// ---------------------------------------------------------------------------
// Launcher
// ---------------------------------------------------------------------------
extern "C" void kernel_launch(void* const* d_in, const int* in_sizes, int n_in,
                              void* d_out, int out_size, void* d_ws, size_t ws_size,
                              hipStream_t stream) {
  (void)in_sizes; (void)n_in; (void)out_size; (void)ws_size;
  const float* x        = (const float*)d_in[0];
  const float* ab       = (const float*)d_in[1];
  const float* qkv_w    = (const float*)d_in[2];
  const float* qkv_b    = (const float*)d_in[3];
  const float* q_ln_w   = (const float*)d_in[4];
  const float* k_ln_w   = (const float*)d_in[5];
  const float* out_w    = (const float*)d_in[6];
  const float* out_b    = (const float*)d_in[7];

  const int B = 2, S = 2048, D = 2048, M = B * S;

  char* ws = (char*)d_ws;
  u16* xb   = (u16*)ws;               ws += (size_t)M * D * 2;        // 16 MB
  u16* wqkv = (u16*)ws;               ws += (size_t)3 * D * D * 2;    // 24 MB
  u16* wout = (u16*)ws;               ws += (size_t)D * D * 2;        //  8 MB
  u16* qkvb = (u16*)ws;               ws += (size_t)M * 3 * D * 2;    // 48 MB
  u16* Qb   = (u16*)ws;               ws += (size_t)M * D * 2;        // 16 MB
  u16* Kb   = (u16*)ws;               ws += (size_t)M * D * 2;        // 16 MB
  u16* Vb   = (u16*)ws;               ws += (size_t)M * D * 2;        // 16 MB
  u16* ctx  = (u16*)ws;               ws += (size_t)M * D * 2;        // 16 MB

  // 1) convert inputs to bf16
  f32_to_bf16_k<<<2048, 256, 0, stream>>>(x, xb, M * D);
  f32_to_bf16_k<<<2048, 256, 0, stream>>>(qkv_w, wqkv, 3 * D * D);
  f32_to_bf16_k<<<2048, 256, 0, stream>>>(out_w, wout, D * D);

  // 2) qkv = x @ qkv_w^T + qkv_b   (bf16 out)
  gemm_bf16_k<true><<<dim3(3 * D / 128, M / 128), 256, 0, stream>>>(
      xb, wqkv, qkv_b, (void*)qkvb, M, 3 * D, D);

  // 3) rmsnorm q/k, split to [B,H,S,d] bf16
  rmsnorm_split_k<<<M, 256, 0, stream>>>(qkvb, q_ln_w, k_ln_w, Qb, Kb, Vb, S, D);

  // 4) flash attention -> ctx [B*S, D] bf16
  const float scale = 1.0f / sqrtf(128.0f);
  attn_k<<<dim3(S / 128, B * 16), 256, 80 * 1024, stream>>>(
      Qb, Kb, Vb, ab, ctx, S, scale);

  // 5) out = ctx @ out_w^T + out_b  (f32 out)
  gemm_bf16_k<false><<<dim3(D / 128, M / 128), 256, 0, stream>>>(
      ctx, wout, out_b, d_out, M, D, D);
}
